// DGCNN_Model_2628519985494
// MI455X (gfx1250) — matrix operand order repaired and checked by reading, not compiled, not run on hardware
//
#include <hip/hip_runtime.h>
#include <hip/hip_bf16.h>

// ---------------------------------------------------------------------------
// DGCNN on gfx1250 (MI455X). All GEMMs (kNN Gram, EdgeConv MLP, conv5) run on
// v_wmma_f32_16x16x32_f16 (one 16x16 tile per wave32). EdgeConv and conv5
// stage their weight tiles into LDS with gfx1250 async loads
// (global_load_async_to_lds_b128, double-buffered, s_wait_asynccnt) shared by
// 8 waves/block; WMMA chains use dual accumulators to keep two matrix ops in
// flight. Pooling / max-over-k fused via agent-scope float atomic max/add.
// ---------------------------------------------------------------------------

typedef __attribute__((ext_vector_type(16))) _Float16 v16h;
typedef __attribute__((ext_vector_type(8)))  _Float16 v8h;
typedef __attribute__((ext_vector_type(8)))  float    v8f;

static constexpr int Bn = 8;
static constexpr int Np = 2048;
static constexpr int Kn = 20;
static constexpr float RSQ_BN = 0.99999500003749977f;   // 1/sqrt(1+1e-5)

// --- helpers ---------------------------------------------------------------

__device__ __forceinline__ v16h load_row16(const _Float16* p) {
  v8h a = *(const v8h*)p;
  v8h b = *(const v8h*)(p + 16);
  v16h r;
#pragma unroll
  for (int t = 0; t < 8; ++t) { r[t] = a[t]; r[t + 8] = b[t]; }
  return r;
}

// LDS byte offset of a __shared__ object (flat LDS aperture: addr[31:0] = offset)
__device__ __forceinline__ uint32_t lds_addr(const void* p) {
  return (uint32_t)(uintptr_t)p;
}

// CDNA5 async copy: global -> LDS, 16 bytes, tracked by ASYNCcnt
__device__ __forceinline__ void async_ld_b128(uint32_t lds_off, const void* gaddr) {
  asm volatile("global_load_async_to_lds_b128 %0, %1, off"
               :: "v"(lds_off), "v"(gaddr) : "memory");
}
__device__ __forceinline__ void wait_async0() {
  asm volatile("s_wait_asynccnt 0x0" ::: "memory");
}

// --- weight prep: f32 -> f16, zero-padded K --------------------------------

__global__ void pad_weights_kernel(const float* __restrict__ w, _Float16* __restrict__ wp,
                                   int Cout, int Cin, int CinPad) {
  int i = blockIdx.x * blockDim.x + threadIdx.x;
  if (i >= Cout * CinPad) return;
  int o = i / CinPad, c = i % CinPad;
  wp[i] = (_Float16)(c < Cin ? w[(size_t)o * Cin + c] : 0.f);
}

// --- x (B,3,N) -> padded f16 (B,N,32) + squared norms ----------------------

__global__ __launch_bounds__(256) void prep_xt_kernel(const float* __restrict__ x,
                                                      _Float16* __restrict__ xtp,
                                                      float* __restrict__ xx) {
  int i = blockIdx.x * 256 + threadIdx.x;          // B*Np points
  int b = i / Np, n = i % Np;
  float s = 0.f;
  _Float16* d = xtp + (size_t)i * 32;
#pragma unroll
  for (int c = 0; c < 32; ++c) {
    float v = 0.f;
    if (c < 3) { v = x[((size_t)b * 3 + c) * Np + n]; s += v * v; }
    d[c] = (_Float16)v;
  }
  xx[i] = s;
}

// --- Gram matrix G[b][i][j] = <X_i, X_j> via WMMA --------------------------
// One wave computes one 16x16 tile. CHUNKS = channels/32.

template<int CHUNKS>
__global__ __launch_bounds__(128) void gram_kernel(const _Float16* __restrict__ X, int rowStride,
                                                   float* __restrict__ G) {
  const int lane  = threadIdx.x & 31;
  const int gwave = blockIdx.x * 4 + (threadIdx.x >> 5);
  const int T = Np / 16;                 // 128 tiles per dim
  const int tilesPerBatch = T * T;
  const int b = gwave / tilesPerBatch;
  const int t = gwave % tilesPerBatch;
  const int i0 = (t / T) * 16, j0 = (t % T) * 16;
  const int m = lane & 15, half = lane >> 4;
  const _Float16* Xb = X + (size_t)b * Np * rowStride;
  v8f c = {};
#pragma unroll
  for (int ch = 0; ch < CHUNKS; ++ch) {
    const _Float16* ar = Xb + (size_t)(i0 + m) * rowStride + ch * 32;
    v8h a0 = *(const v8h*)(ar + half * 8);
    v8h a1 = *(const v8h*)(ar + 16 + half * 8);
    v16h A;
#pragma unroll
    for (int e = 0; e < 8; ++e) { A[e] = a0[e]; A[e + 8] = a1[e]; }
    v16h Bf = load_row16(Xb + (size_t)(j0 + m) * rowStride + ch * 32 + half * 8);
    c = __builtin_amdgcn_wmma_f32_16x16x32_f16(false, A, false, Bf, (short)0, c, false, false);
  }
  float* g = G + ((size_t)b * Np + i0 + half * 8) * Np + j0 + m;
#pragma unroll
  for (int d = 0; d < 8; ++d) g[(size_t)d * Np] = c[d];
}

// --- top-k=20 neighbor indices from pd = 2G - xx_i - xx_j ------------------

__global__ __launch_bounds__(256) void topk_kernel(const float* __restrict__ G,
                                                   const float* __restrict__ xx,
                                                   int* __restrict__ nidx) {
  int i = blockIdx.x * 256 + threadIdx.x;          // row id, B*Np total (exact grid)
  const int b = i / Np;
  const float* Grow = G + (size_t)i * Np;
  const float* xxb  = xx + (size_t)b * Np;
  const float xi = xx[i];
  float bv[Kn]; int bi[Kn];
#pragma unroll
  for (int s = 0; s < Kn; ++s) { bv[s] = -3.0e38f; bi[s] = 0; }
  for (int j = 0; j < Np; ++j) {
    float pd = 2.f * Grow[j] - xi - xxb[j];
    if (pd > bv[Kn - 1]) {
      float v = pd; int jj = j;
#pragma unroll
      for (int s = 0; s < Kn; ++s) {
        bool gt = v > bv[s];
        float tv = bv[s]; int ti = bi[s];
        if (gt) { bv[s] = v; bi[s] = jj; v = tv; jj = ti; }
      }
    }
  }
  int* dst = nidx + (size_t)i * Kn;
#pragma unroll
  for (int s = 0; s < Kn; ++s) dst[s] = bi[s];
}

// --- generic fill ----------------------------------------------------------

__global__ void fill_kernel(float* p, float v, int n) {
  int i = blockIdx.x * blockDim.x + threadIdx.x;
  if (i < n) p[i] = v;
}

// --- EdgeConv: gather + (N*K, 2C) x (2C, Cout) WMMA GEMM + BN + LReLU ------
// + fused max-over-k via float atomic max. 8 waves/block, one 16-edge row
// tile per wave; the 16xCinPad weight tile for each column tile is staged
// into LDS once per block via async copies (double-buffered) and shared.

template<int C, int CHUNKS, int COLT>
__global__ __launch_bounds__(256) void edgeconv_kernel(const _Float16* __restrict__ X, int rowStride,
                                                       const int* __restrict__ nidx,
                                                       const _Float16* __restrict__ W,
                                                       const float* __restrict__ gamma,
                                                       const float* __restrict__ beta,
                                                       float* __restrict__ outMax) {
  constexpr int CinPad = CHUNKS * 32;
  constexpr int TILE   = 16 * CinPad;              // halves per weight tile
  constexpr int XFERS  = TILE / 8;                 // b128 copies per tile
  __shared__ _Float16 wtile[2][TILE];

  const int tid   = threadIdx.x;
  const int lane  = tid & 31;
  const int gwave = blockIdx.x * 8 + (tid >> 5);
  const int rowTiles = Np * Kn / 16;               // 2560 per batch
  const int b  = gwave / rowTiles;
  const int rt = gwave % rowTiles;
  const int m = lane & 15, half = lane >> 4;
  const int e = rt * 16 + m;                       // edge id within batch
  const int n = e / Kn, kk = e % Kn;
  const int j = nidx[((size_t)b * Np + n) * Kn + kk];
  const _Float16* xn = X + ((size_t)b * Np + n) * rowStride;
  const _Float16* xj = X + ((size_t)b * Np + j) * rowStride;

  v16h A[CHUNKS];
#pragma unroll
  for (int ch = 0; ch < CHUNKS; ++ch) {
    const int kg0 = ch * 32 + half * 8;
    const int kg1 = ch * 32 + 16 + half * 8;
    if constexpr (C == 3) {                        // layer 1: Cin=6 padded to 32
#pragma unroll
      for (int t = 0; t < 8; ++t) {
        int k0 = kg0 + t, k1 = kg1 + t;
        _Float16 v0 = (k0 < 3) ? (_Float16)(xj[k0] - xn[k0])
                               : ((k0 < 6) ? xn[k0 - 3] : (_Float16)0.f);
        _Float16 v1 = (k1 < 3) ? (_Float16)(xj[k1] - xn[k1])
                               : ((k1 < 6) ? xn[k1 - 3] : (_Float16)0.f);
        A[ch][t]     = v0;
        A[ch][t + 8] = v1;
      }
    } else {                                       // C multiple of 8: contiguous groups
      if (kg0 < C) {
        v8h aj = *(const v8h*)(xj + kg0); v8h an = *(const v8h*)(xn + kg0);
#pragma unroll
        for (int t = 0; t < 8; ++t) A[ch][t] = aj[t] - an[t];   // f16 sub -> v_pk_sub
      } else {
        v8h an = *(const v8h*)(xn + kg0 - C);
#pragma unroll
        for (int t = 0; t < 8; ++t) A[ch][t] = an[t];
      }
      if (kg1 < C) {
        v8h aj = *(const v8h*)(xj + kg1); v8h an = *(const v8h*)(xn + kg1);
#pragma unroll
        for (int t = 0; t < 8; ++t) A[ch][t + 8] = aj[t] - an[t];
      } else {
        v8h an = *(const v8h*)(xn + kg1 - C);
#pragma unroll
        for (int t = 0; t < 8; ++t) A[ch][t + 8] = an[t];
      }
    }
  }

  // stage weight tile 0 (rows 0..15 of W, contiguous)
  for (int t = tid; t < XFERS; t += 256)
    async_ld_b128(lds_addr(&wtile[0][t * 8]), W + t * 8);
  wait_async0();
  __syncthreads();

  const int Cout = COLT * 16;
#pragma unroll 1
  for (int ct = 0; ct < COLT; ++ct) {
    const int cur = ct & 1;
    if (ct + 1 < COLT) {                           // stage next tile into other buffer
      const _Float16* gsrc = W + (size_t)(ct + 1) * TILE;
      for (int t = tid; t < XFERS; t += 256)
        async_ld_b128(lds_addr(&wtile[cur ^ 1][t * 8]), gsrc + t * 8);
    }
    const int c = ct * 16 + m;                     // output channel (D: N = lane&15)
    const _Float16* Wc = &wtile[cur][m * CinPad + half * 8];
    v8f acc = {};
    if constexpr (CHUNKS >= 2) {                   // dual accumulators
      v8f acc1 = {};
#pragma unroll
      for (int ch = 0; ch < CHUNKS; ch += 2) {
        v16h B0 = load_row16(Wc + ch * 32);
        v16h B1 = load_row16(Wc + (ch + 1) * 32);
        acc  = __builtin_amdgcn_wmma_f32_16x16x32_f16(false, A[ch],     false, B0, (short)0, acc,  false, false);
        acc1 = __builtin_amdgcn_wmma_f32_16x16x32_f16(false, A[ch + 1], false, B1, (short)0, acc1, false, false);
      }
#pragma unroll
      for (int d2 = 0; d2 < 8; ++d2) acc[d2] += acc1[d2];
    } else {
      v16h B0 = load_row16(Wc);
      acc = __builtin_amdgcn_wmma_f32_16x16x32_f16(false, A[0], false, B0, (short)0, acc, false, false);
    }
    const float gc = gamma[c] * RSQ_BN;
    const float bc = beta[c];
#pragma unroll
    for (int d = 0; d < 8; ++d) {                  // D row = half*8 + d
      float v = acc[d] * gc + bc;
      v = v >= 0.f ? v : 0.2f * v;
      const int ed = rt * 16 + half * 8 + d;
      const int nd = ed / Kn;
      float* p = outMax + ((size_t)b * Np + nd) * Cout + c;
      __hip_atomic_fetch_max(p, v, __ATOMIC_RELAXED, __HIP_MEMORY_SCOPE_AGENT);
    }
    wait_async0();                                 // own async copies done
    __syncthreads();                               // tile ct+1 visible to all waves
  }
}

// --- finalize a layer: maxed f32 -> f16 slice of cat buffer + new norms ----

__global__ __launch_bounds__(256) void finalize_kernel(const float* __restrict__ src,
                                                       _Float16* __restrict__ catb,
                                                       float* __restrict__ xx,
                                                       int Cout, int chOff) {
  int i = blockIdx.x * 256 + threadIdx.x;          // B*Np points
  const float* s = src + (size_t)i * Cout;
  _Float16* d = catb + (size_t)i * 512 + chOff;
  float acc = 0.f;
  for (int c = 0; c < Cout; ++c) { float v = s[c]; acc += v * v; d[c] = (_Float16)v; }
  xx[i] = acc;
}

// --- conv5: (B*N,512)x(512,1024) WMMA + BN + LReLU + fused max/mean pool ---
// 8 waves / block; the 16x512 weight tile for each column tile is staged into
// LDS once per block with async global->LDS copies (double-buffered) and
// shared by all waves. A fragments (full K=512) live in registers.

__global__ __launch_bounds__(256) void conv5_kernel(const _Float16* __restrict__ cat,
                                                    const _Float16* __restrict__ W5,
                                                    const float* __restrict__ g5,
                                                    const float* __restrict__ b5,
                                                    float* __restrict__ hmax,
                                                    float* __restrict__ hsum) {
  __shared__ _Float16 wtile[2][16 * 512];          // 2 x 16KB double buffer
  const int tid  = threadIdx.x;
  const int lane = tid & 31;
  const int rt = blockIdx.x * 8 + (tid >> 5);      // 1024 row tiles, 8 waves/block
  const int m = lane & 15, half = lane >> 4;
  const int r = rt * 16 + m;
  const _Float16* ar = cat + (size_t)r * 512;
  v16h A[16];
#pragma unroll
  for (int ch = 0; ch < 16; ++ch) {
    v8h a0 = *(const v8h*)(ar + ch * 32 + half * 8);
    v8h a1 = *(const v8h*)(ar + ch * 32 + 16 + half * 8);
#pragma unroll
    for (int t = 0; t < 8; ++t) { A[ch][t] = a0[t]; A[ch][t + 8] = a1[t]; }
  }
  const int b = (rt * 16) / Np;                    // uniform per tile (2048 % 16 == 0)

  // stage column tile 0: 8192 halves, 32 per thread, 4 async b128 each
  {
    const _Float16* gsrc = W5 + (size_t)tid * 32;
    _Float16* ldst = &wtile[0][tid * 32];
#pragma unroll
    for (int t = 0; t < 4; ++t)
      async_ld_b128(lds_addr(ldst + t * 8), gsrc + t * 8);
  }
  wait_async0();
  __syncthreads();

#pragma unroll 1
  for (int ct = 0; ct < 64; ++ct) {
    const int cur = ct & 1;
    if (ct + 1 < 64) {                             // stage next tile into other buffer
      const _Float16* gsrc = W5 + (size_t)(ct + 1) * 16 * 512 + tid * 32;
      _Float16* ldst = &wtile[cur ^ 1][tid * 32];
#pragma unroll
      for (int t = 0; t < 4; ++t)
        async_ld_b128(lds_addr(ldst + t * 8), gsrc + t * 8);
    }
    const int c = ct * 16 + m;
    const _Float16* Wc = &wtile[cur][m * 512 + half * 8];
    v8f acc = {}, acc1 = {};
#pragma unroll
    for (int ch = 0; ch < 16; ch += 2) {           // dual accumulators
      v16h B0 = load_row16(Wc + ch * 32);
      v16h B1 = load_row16(Wc + (ch + 1) * 32);
      acc  = __builtin_amdgcn_wmma_f32_16x16x32_f16(false, A[ch],     false, B0, (short)0, acc,  false, false);
      acc1 = __builtin_amdgcn_wmma_f32_16x16x32_f16(false, A[ch + 1], false, B1, (short)0, acc1, false, false);
    }
#pragma unroll
    for (int d2 = 0; d2 < 8; ++d2) acc[d2] += acc1[d2];
    const float gc = g5[c] * RSQ_BN;
    const float bc = b5[c];
    float lmax = -3.0e38f, lsum = 0.f;
#pragma unroll
    for (int d = 0; d < 8; ++d) {
      float v = acc[d] * gc + bc;
      v = v >= 0.f ? v : 0.2f * v;
      lmax = fmaxf(lmax, v);
      lsum += v;
    }
    float* pm = hmax + (size_t)b * 1024 + c;
    float* ps = hsum + (size_t)b * 1024 + c;
    __hip_atomic_fetch_max(pm, lmax, __ATOMIC_RELAXED, __HIP_MEMORY_SCOPE_AGENT);
    __hip_atomic_fetch_add(ps, lsum, __ATOMIC_RELAXED, __HIP_MEMORY_SCOPE_AGENT);
    wait_async0();                                 // own async copies done
    __syncthreads();                               // tile ct+1 visible to all waves
  }
}

// --- FC head: z(2048) -> 512 -> 256 -> 40, tiny, LDS-staged VALU -----------

__global__ __launch_bounds__(256) void fc_head_kernel(const float* __restrict__ hmax,
                                                      const float* __restrict__ hsum,
                                                      const float* __restrict__ fc1w,
                                                      const float* __restrict__ g6,
                                                      const float* __restrict__ b6,
                                                      const float* __restrict__ fc2w,
                                                      const float* __restrict__ g7,
                                                      const float* __restrict__ b7,
                                                      const float* __restrict__ fc3w,
                                                      const float* __restrict__ fc3b,
                                                      float* __restrict__ out) {
  __shared__ float z[2048];
  __shared__ float z1[512];
  __shared__ float z2[256];
  const int b = blockIdx.x, t = threadIdx.x;
  for (int c = t; c < 1024; c += 256) {
    z[c]        = hmax[(size_t)b * 1024 + c];
    z[1024 + c] = hsum[(size_t)b * 1024 + c] * (1.f / (float)Np);
  }
  __syncthreads();
  for (int o = t; o < 512; o += 256) {
    float s = 0.f;
    const float* wr = fc1w + (size_t)o * 2048;
    for (int c = 0; c < 2048; ++c) s += z[c] * wr[c];
    s = s * RSQ_BN * g6[o] + b6[o];
    z1[o] = s >= 0.f ? s : 0.2f * s;
  }
  __syncthreads();
  for (int o = t; o < 256; o += 256) {
    float s = 0.f;
    const float* wr = fc2w + (size_t)o * 512;
    for (int c = 0; c < 512; ++c) s += z1[c] * wr[c];
    s = s * RSQ_BN * g7[o] + b7[o];
    z2[o] = s >= 0.f ? s : 0.2f * s;
  }
  __syncthreads();
  if (t < 40) {
    float s = 0.f;
    const float* wr = fc3w + (size_t)t * 256;
    for (int c = 0; c < 256; ++c) s += z2[c] * wr[c];
    out[(size_t)b * 40 + t] = s + fc3b[t];
  }
}

// ---------------------------------------------------------------------------

extern "C" void kernel_launch(void* const* d_in, const int* in_sizes, int n_in,
                              void* d_out, int out_size, void* d_ws, size_t ws_size,
                              hipStream_t stream) {
  (void)in_sizes; (void)n_in; (void)out_size; (void)ws_size;
  const float* x    = (const float*)d_in[0];
  const float* w1   = (const float*)d_in[1];
  const float* g1   = (const float*)d_in[2];
  const float* b1   = (const float*)d_in[3];
  const float* w2   = (const float*)d_in[4];
  const float* g2   = (const float*)d_in[5];
  const float* b2   = (const float*)d_in[6];
  const float* w3   = (const float*)d_in[7];
  const float* g3   = (const float*)d_in[8];
  const float* b3   = (const float*)d_in[9];
  const float* w4   = (const float*)d_in[10];
  const float* g4   = (const float*)d_in[11];
  const float* b4   = (const float*)d_in[12];
  const float* w5   = (const float*)d_in[13];
  const float* g5   = (const float*)d_in[14];
  const float* b5   = (const float*)d_in[15];
  const float* fc1w = (const float*)d_in[16];
  const float* g6   = (const float*)d_in[17];
  const float* b6   = (const float*)d_in[18];
  const float* fc2w = (const float*)d_in[19];
  const float* g7   = (const float*)d_in[20];
  const float* b7   = (const float*)d_in[21];
  const float* fc3w = (const float*)d_in[22];
  const float* fc3b = (const float*)d_in[23];
  float* out = (float*)d_out;

  // workspace carve-up (~171 MB total)
  char* ws = (char*)d_ws;
  size_t off = 0;
  auto take = [&](size_t bytes) -> char* {
    char* p = ws + off;
    off = (off + bytes + 255) & ~(size_t)255;
    return p;
  };
  float*    G    = (float*)take((size_t)Bn * Np * Np * 4);        // 134 MB (L2-resident)
  int*      nidx = (int*)take((size_t)Bn * Np * Kn * 4);
  _Float16* xtp  = (_Float16*)take((size_t)Bn * Np * 32 * 2);
  _Float16* catb = (_Float16*)take((size_t)Bn * Np * 512 * 2);
  float*    xx   = (float*)take((size_t)Bn * Np * 4);
  float*    om   = (float*)take((size_t)Bn * Np * 256 * 4);
  _Float16* w1p  = (_Float16*)take((size_t)64 * 32 * 2);
  _Float16* w2p  = (_Float16*)take((size_t)64 * 128 * 2);
  _Float16* w3p  = (_Float16*)take((size_t)128 * 128 * 2);
  _Float16* w4p  = (_Float16*)take((size_t)256 * 256 * 2);
  _Float16* w5p  = (_Float16*)take((size_t)1024 * 512 * 2);
  float*    hmax = (float*)take((size_t)Bn * 1024 * 4);
  float*    hsum = (float*)take((size_t)Bn * 1024 * 4);

  const int PTS = Bn * Np;                         // 16384
  const int gramBlocks = Bn * (Np / 16) * (Np / 16) / 4;  // 32768
  const int ecBlocks   = Bn * (Np * Kn / 16) / 8;         // 2560 (8 waves/block)

  // weights -> padded f16
  pad_weights_kernel<<<(64 * 32 + 255) / 256, 256, 0, stream>>>(w1, w1p, 64, 6, 32);
  pad_weights_kernel<<<(64 * 128 + 255) / 256, 256, 0, stream>>>(w2, w2p, 64, 128, 128);
  pad_weights_kernel<<<(128 * 128 + 255) / 256, 256, 0, stream>>>(w3, w3p, 128, 128, 128);
  pad_weights_kernel<<<(256 * 256 + 255) / 256, 256, 0, stream>>>(w4, w4p, 256, 256, 256);
  pad_weights_kernel<<<(1024 * 512 + 255) / 256, 256, 0, stream>>>(w5, w5p, 1024, 512, 512);
  prep_xt_kernel<<<PTS / 256, 256, 0, stream>>>(x, xtp, xx);

  // ---- EdgeConv layer 1: C=3 (pad 32), Cout=64 ----
  gram_kernel<1><<<gramBlocks, 128, 0, stream>>>(xtp, 32, G);
  topk_kernel<<<PTS / 256, 256, 0, stream>>>(G, xx, nidx);
  fill_kernel<<<(PTS * 64 + 255) / 256, 256, 0, stream>>>(om, -3.0e38f, PTS * 64);
  edgeconv_kernel<3, 1, 4><<<ecBlocks, 256, 0, stream>>>(xtp, 32, nidx, w1p, g1, b1, om);
  finalize_kernel<<<PTS / 256, 256, 0, stream>>>(om, catb, xx, 64, 0);

  // ---- EdgeConv layer 2: C=64, Cout=64 ----
  gram_kernel<2><<<gramBlocks, 128, 0, stream>>>(catb + 0, 512, G);
  topk_kernel<<<PTS / 256, 256, 0, stream>>>(G, xx, nidx);
  fill_kernel<<<(PTS * 64 + 255) / 256, 256, 0, stream>>>(om, -3.0e38f, PTS * 64);
  edgeconv_kernel<64, 4, 4><<<ecBlocks, 256, 0, stream>>>(catb + 0, 512, nidx, w2p, g2, b2, om);
  finalize_kernel<<<PTS / 256, 256, 0, stream>>>(om, catb, xx, 64, 64);

  // ---- EdgeConv layer 3: C=64, Cout=128 ----
  gram_kernel<2><<<gramBlocks, 128, 0, stream>>>(catb + 64, 512, G);
  topk_kernel<<<PTS / 256, 256, 0, stream>>>(G, xx, nidx);
  fill_kernel<<<(PTS * 128 + 255) / 256, 256, 0, stream>>>(om, -3.0e38f, PTS * 128);
  edgeconv_kernel<64, 4, 8><<<ecBlocks, 256, 0, stream>>>(catb + 64, 512, nidx, w3p, g3, b3, om);
  finalize_kernel<<<PTS / 256, 256, 0, stream>>>(om, catb, xx, 128, 128);

  // ---- EdgeConv layer 4: C=128, Cout=256 ----
  gram_kernel<4><<<gramBlocks, 128, 0, stream>>>(catb + 128, 512, G);
  topk_kernel<<<PTS / 256, 256, 0, stream>>>(G, xx, nidx);
  fill_kernel<<<(PTS * 256 + 255) / 256, 256, 0, stream>>>(om, -3.0e38f, PTS * 256);
  edgeconv_kernel<128, 8, 16><<<ecBlocks, 256, 0, stream>>>(catb + 128, 512, nidx, w4p, g4, b4, om);
  finalize_kernel<<<PTS / 256, 256, 0, stream>>>(om, catb, xx, 256, 256);

  // ---- conv5 (512 -> 1024) + fused global max/mean pooling ----
  fill_kernel<<<(Bn * 1024 + 255) / 256, 256, 0, stream>>>(hmax, -3.0e38f, Bn * 1024);
  fill_kernel<<<(Bn * 1024 + 255) / 256, 256, 0, stream>>>(hsum, 0.f, Bn * 1024);
  conv5_kernel<<<(PTS / 16) / 8, 256, 0, stream>>>(catb, w5p, g5, b5, hmax, hsum);

  // ---- FC head ----
  fc_head_kernel<<<Bn, 256, 0, stream>>>(hmax, hsum, fc1w, g6, b6, fc2w, g7, b7, fc3w, fc3b, out);
}
